// xDeepFM_Attention_52853867544736
// MI455X (gfx1250) — matrix operand order repaired and checked by reading, not compiled, not run on hardware
//
#include <hip/hip_runtime.h>

// ---------------------------------------------------------------------------
// xDeepFM-Attention forward for MI455X (gfx1250, wave32, WMMA bf16->f32)
// ---------------------------------------------------------------------------

typedef __attribute__((ext_vector_type(16))) __bf16 v16bf;
typedef __attribute__((ext_vector_type(8)))  float  v8f;
typedef __attribute__((ext_vector_type(4)))  unsigned int v4u;

#define B_SZ    1024
#define NDENSE  13
#define NSPARSE 26
#define DFEAT   39
#define VOCAB   10000
#define KDIM    16
#define CINH    128      // CIN hidden width per layer
#define DIN     429      // 13 + 26*16
#define DPAD    448      // DIN padded to 14*32
#define HID     400
#define HPAD    416      // HID padded to 13*32

union BF16x16 { v16bf v; unsigned short u[16]; unsigned int w[8]; v4u q[2]; };

static __device__ __forceinline__ unsigned short f2bf(float f) {
    // round-to-nearest-even float -> bf16 bits
    unsigned int x = __float_as_uint(f);
    unsigned int r = x + 0x7FFFu + ((x >> 16) & 1u);
    return (unsigned short)(r >> 16);
}

// ---------------------------------------------------------------------------
// Pack an fp32 weight matrix [G*Pg rows x H cols] into WMMA B-fragment tiles.
// Tile id = ((g*stepsPerGroup + s)*ntiles + nt); each tile = 512 bf16,
// layout: lane*16 + j, with K = s*32 + (j<8 ? 0 : 16) + (lane>=16 ? 8 : 0) + (j&7), N = nt*16+lane%16.
// Rows beyond Pg within a group are zero-padded (CIN f-groups, MLP K padding).
// ---------------------------------------------------------------------------
__global__ __launch_bounds__(32) void pack_b_tiles(
    const float* __restrict__ src, unsigned short* __restrict__ dst,
    int G, int Pg, int H, int stepsPerGroup, int ntiles)
{
    int tileId = blockIdx.x;
    int nt = tileId % ntiles;
    int s  = (tileId / ntiles) % stepsPerGroup;
    int g  = tileId / (ntiles * stepsPerGroup);
    int lane = threadIdx.x;
    int hi = lane >> 4, lm = lane & 15;
    int n = nt * 16 + lm;
    unsigned short* out = dst + (size_t)tileId * 512 + lane * 16;
    for (int j = 0; j < 16; ++j) {
        int kk = s * 32 + (j < 8 ? 0 : 16) + hi * 8 + (j & 7);
        float v = 0.f;
        if (kk < Pg && n < H) v = src[(size_t)(g * Pg + kk) * H + n];
        out[j] = f2bf(v);
    }
}

// ---------------------------------------------------------------------------
// BN + embedding gather + linear head + MLP input assembly (bf16, K-padded)
// ---------------------------------------------------------------------------
__global__ __launch_bounds__(64) void embed_bn_kernel(
    const float* __restrict__ dense_x, const int* __restrict__ sparse_idx,
    const float* __restrict__ gamma, const float* __restrict__ beta,
    const float* __restrict__ bn_mean, const float* __restrict__ bn_var,
    const float* __restrict__ tables, const float* __restrict__ w_lin,
    const float* __restrict__ b_lin,
    float* __restrict__ emb, unsigned short* __restrict__ h0,
    float* __restrict__ linear_out)
{
    __shared__ float s_xh[DFEAT];
    __shared__ int   s_idx[NSPARSE];
    int b = blockIdx.x, t = threadIdx.x;
    if (t < DFEAT) {
        float x = (t < NDENSE) ? dense_x[b * NDENSE + t]
                               : (float)sparse_idx[b * NSPARSE + (t - NDENSE)];
        s_xh[t] = (x - bn_mean[t]) * rsqrtf(bn_var[t] + 1e-3f) * gamma[t] + beta[t];
    }
    __syncthreads();
    if (t == 0) {
        float acc = b_lin[0];
        for (int d = 0; d < DFEAT; ++d) acc += s_xh[d] * w_lin[d];
        linear_out[b] = acc;
    }
    if (t < NSPARSE) {
        int ix = (int)s_xh[NDENSE + t];           // trunc toward zero, like astype
        ix = ix < 0 ? 0 : (ix > VOCAB - 1 ? VOCAB - 1 : ix);
        s_idx[t] = ix;
    }
    __syncthreads();
    for (int i = t; i < NSPARSE * KDIM; i += 64) {
        int j = i >> 4, kk = i & 15;
        float v = tables[((size_t)j * VOCAB + s_idx[j]) * KDIM + kk];
        emb[(size_t)b * (NSPARSE * KDIM) + i] = v;
        h0[(size_t)b * DPAD + NDENSE + i] = f2bf(v);
    }
    if (t < NDENSE) h0[(size_t)b * DPAD + t] = f2bf(s_xh[t]);
    for (int i = DIN + t; i < DPAD; i += 64) h0[(size_t)b * DPAD + i] = 0;  // K pad
}

// ---------------------------------------------------------------------------
// CIN layer 1: fi = 26.  out[b][k,h] = sum_{f,g} softmax_f(u_f v_g)*X0v_f * cw0[f*26+g,h]
// One workgroup (8 wave32) per batch row; wave w owns n-tile w (h = 16w..16w+15).
// Ping-pong A tile -> one barrier per f-group.
// ---------------------------------------------------------------------------
__global__ __launch_bounds__(256) void cin_layer1(
    const float* __restrict__ emb,
    const float* __restrict__ wq, const float* __restrict__ wk,
    const float* __restrict__ wv,
    const unsigned short* __restrict__ cwp,   // packed [26][1][8][512]
    float* __restrict__ r1)
{
    __shared__ float s_X0k[416], s_X0v[416], s_Xq[416];
    __shared__ float s_mx[416], s_rden[416];         // index g*16+k (rden = 1/den)
    __shared__ __align__(16) unsigned short s_A[2][512];  // [k][g] 16x32 bf16, ping-pong
    int b = blockIdx.x, t = threadIdx.x;
    for (int i = t; i < 416; i += 256) {
        float e = emb[(size_t)b * 416 + i];
        s_X0k[i] = e * wk[i];
        s_X0v[i] = e * wv[i];
        s_Xq[i]  = e * wq[i];
    }
    __syncthreads();
    for (int p = t; p < 416; p += 256) {
        int k = p & 15;
        float q = s_Xq[p];
        float m = -3.4e38f;
        for (int f = 0; f < 26; ++f) { float v = s_X0k[f * 16 + k] * q; m = v > m ? v : m; }
        float den = 0.f;
        for (int f = 0; f < 26; ++f) den += __expf(s_X0k[f * 16 + k] * q - m);
        s_mx[p] = m; s_rden[p] = 1.0f / den;
    }
    __syncthreads();
    int wave = t >> 5, lane = t & 31, hi = lane >> 4, lm = lane & 15;
    v8f acc = {};
    for (int f = 0; f < 26; ++f) {
        unsigned short* Ab = s_A[f & 1];
        // build A tile: 512 entries, exactly 2 per thread, no guards
#pragma unroll
        for (int it = 0; it < 2; ++it) {
            int idx = t + it * 256;
            int k = idx >> 5, g = idx & 31;
            float val = 0.f;
            if (g < 26) {
                int p = g * 16 + k;
                float v = s_X0k[f * 16 + k] * s_Xq[p];
                val = __expf(v - s_mx[p]) * s_rden[p] * s_X0v[f * 16 + k];
            }
            Ab[idx] = f2bf(val);
        }
        __syncthreads();
        BF16x16 a, bb;
        int k0 = hi * 8;
        a.q[0] = *(const v4u*)&Ab[lm * 32 + k0];
        a.q[1] = *(const v4u*)&Ab[lm * 32 + 16 + k0];
        bb.v = *(const v16bf*)(cwp + ((size_t)(f * 8 + wave)) * 512 + lane * 16);
        acc = __builtin_amdgcn_wmma_f32_16x16x32_bf16(false, a.v, false, bb.v,
                                                      (short)0, acc, false, false);
    }
    int h = wave * 16 + lm;
#pragma unroll
    for (int r = 0; r < 8; ++r)
        r1[(size_t)b * 2048 + h * 16 + hi * 8 + r] = acc[r];
}

// ---------------------------------------------------------------------------
// CIN layer 2: fi = 128.  Per f: A tile 16x128 (4 K-steps of 32), ping-pong.
// ---------------------------------------------------------------------------
__global__ __launch_bounds__(256) void cin_layer2(
    const float* __restrict__ emb, const float* __restrict__ r1,
    const float* __restrict__ wq, const float* __restrict__ wk,
    const float* __restrict__ wv,
    const unsigned short* __restrict__ cwp,   // packed [26][4][8][512]
    float* __restrict__ r2)
{
    __shared__ float s_X0k[416], s_X0v[416];
    __shared__ float s_Xq[2048], s_mx[2048], s_rden[2048];  // index g*16+k
    __shared__ __align__(16) unsigned short s_A[2][2048];   // [k][g] 16x128 bf16
    int b = blockIdx.x, t = threadIdx.x;
    for (int i = t; i < 416; i += 256) {
        float e = emb[(size_t)b * 416 + i];
        s_X0k[i] = e * wk[i];
        s_X0v[i] = e * wv[i];
    }
#pragma unroll
    for (int it = 0; it < 8; ++it) {
        int i = t + it * 256;
        s_Xq[i] = r1[(size_t)b * 2048 + i] * wq[i];
    }
    __syncthreads();
#pragma unroll
    for (int it = 0; it < 8; ++it) {
        int p = t + it * 256;
        int k = p & 15;
        float q = s_Xq[p];
        float m = -3.4e38f;
        for (int f = 0; f < 26; ++f) { float v = s_X0k[f * 16 + k] * q; m = v > m ? v : m; }
        float den = 0.f;
        for (int f = 0; f < 26; ++f) den += __expf(s_X0k[f * 16 + k] * q - m);
        s_mx[p] = m; s_rden[p] = 1.0f / den;
    }
    __syncthreads();
    int wave = t >> 5, lane = t & 31, hi = lane >> 4, lm = lane & 15;
    v8f acc = {};
    for (int f = 0; f < 26; ++f) {
        unsigned short* Ab = s_A[f & 1];
        // build A tile: 2048 entries, exactly 8 per thread, no guards
#pragma unroll
        for (int it = 0; it < 8; ++it) {
            int idx = t + it * 256;
            int k = idx >> 7, g = idx & 127;
            int p = g * 16 + k;
            float v = s_X0k[f * 16 + k] * s_Xq[p];
            Ab[idx] = f2bf(__expf(v - s_mx[p]) * s_rden[p] * s_X0v[f * 16 + k]);
        }
        __syncthreads();
#pragma unroll
        for (int s = 0; s < 4; ++s) {
            BF16x16 a, bb;
            int k0 = s * 32 + hi * 8;
            a.q[0] = *(const v4u*)&Ab[lm * 128 + k0];
            a.q[1] = *(const v4u*)&Ab[lm * 128 + 16 + k0];
            bb.v = *(const v16bf*)(cwp + ((size_t)((f * 4 + s) * 8 + wave)) * 512 + lane * 16);
            acc = __builtin_amdgcn_wmma_f32_16x16x32_bf16(false, a.v, false, bb.v,
                                                          (short)0, acc, false, false);
        }
    }
    int h = wave * 16 + lm;
#pragma unroll
    for (int r = 0; r < 8; ++r)
        r2[(size_t)b * 2048 + h * 16 + hi * 8 + r] = acc[r];
}

// ---------------------------------------------------------------------------
// cin_out[b][h'] = sum_k r{1,2}[b][h'][k]
// ---------------------------------------------------------------------------
__global__ __launch_bounds__(256) void cin_sum(
    const float* __restrict__ r1, const float* __restrict__ r2,
    float* __restrict__ cin_out)
{
    int b = blockIdx.x, t = threadIdx.x;
    const float* src = (t < 128) ? (r1 + (size_t)b * 2048 + t * 16)
                                 : (r2 + (size_t)b * 2048 + (t - 128) * 16);
    float s = 0.f;
#pragma unroll
    for (int k = 0; k < 16; ++k) s += src[k];
    cin_out[(size_t)b * 256 + t] = s;
}

// ---------------------------------------------------------------------------
// Generic WMMA GEMM: C[16x16 tile] = A[M x steps*32] * Bpacked + bias, opt relu.
// One wave per output tile. A stored as bf16 bits (row-major, lda elements).
// Both A-fragment halves are 16B aligned -> two global_load_b128 per step.
// ---------------------------------------------------------------------------
__global__ __launch_bounds__(32) void gemm_bf16_wmma(
    const unsigned short* __restrict__ A, int lda,
    const unsigned short* __restrict__ Bp,
    const float* __restrict__ bias,
    float* __restrict__ Cf, int ldcf,
    unsigned short* __restrict__ Cb, int ldcb,
    int steps, int ntiles, int relu)
{
    int lane = threadIdx.x;
    int hi = lane >> 4, lm = lane & 15;
    int row0 = blockIdx.x * 16;
    int nt = blockIdx.y;
    const unsigned short* arow = A + (size_t)(row0 + lm) * lda;
    v8f acc = {};
    for (int s = 0; s < steps; ++s) {
        BF16x16 a, b;
        int k0 = s * 32 + hi * 8;
        a.q[0] = *(const v4u*)(arow + k0);
        a.q[1] = *(const v4u*)(arow + k0 + 16);
        b.v = *(const v16bf*)(Bp + (size_t)(s * ntiles + nt) * 512 + lane * 16);
        if (s + 1 < steps)
            __builtin_prefetch(Bp + (size_t)((s + 1) * ntiles + nt) * 512 + lane * 16, 0, 0);
        acc = __builtin_amdgcn_wmma_f32_16x16x32_bf16(false, a.v, false, b.v,
                                                      (short)0, acc, false, false);
    }
    int n = nt * 16 + lm;
#pragma unroll
    for (int r = 0; r < 8; ++r) {
        int m = row0 + hi * 8 + r;
        float v = acc[r] + bias[n];
        if (relu) v = v > 0.f ? v : 0.f;
        if (Cf) Cf[(size_t)m * ldcf + n] = v;
        if (Cb) Cb[(size_t)m * ldcb + n] = f2bf(v);
    }
}

// Zero the K-padding columns of a bf16 activation buffer (needed every call).
__global__ void zero_pad_cols(unsigned short* __restrict__ buf, int ld, int colStart) {
    int i = blockIdx.x * blockDim.x + threadIdx.x;
    int cols = ld - colStart;
    if (i < B_SZ * cols) buf[(size_t)(i / cols) * ld + colStart + (i % cols)] = 0;
}

// ---------------------------------------------------------------------------
// dense_out + final head: pre = sum_h (lin + cin[h] + dout)*wo[h] + bo; sigmoid.
// ---------------------------------------------------------------------------
__global__ __launch_bounds__(256) void final_kernel(
    const float* __restrict__ h3, const float* __restrict__ W4,
    const float* __restrict__ b4, const float* __restrict__ linear_out,
    const float* __restrict__ cin_out, const float* __restrict__ wo,
    const float* __restrict__ bo, float* __restrict__ out)
{
    __shared__ float red[256];
    __shared__ float s_dout;
    int b = blockIdx.x, t = threadIdx.x;
    float p = 0.f;
    for (int j = t; j < HID; j += 256) p += h3[(size_t)b * HID + j] * W4[j];
    red[t] = p; __syncthreads();
    for (int off = 128; off > 0; off >>= 1) {
        if (t < off) red[t] += red[t + off];
        __syncthreads();
    }
    if (t == 0) s_dout = red[0] + b4[0];
    __syncthreads();
    float q = (linear_out[b] + s_dout + cin_out[(size_t)b * 256 + t]) * wo[t];
    __syncthreads();
    red[t] = q; __syncthreads();
    for (int off = 128; off > 0; off >>= 1) {
        if (t < off) red[t] += red[t + off];
        __syncthreads();
    }
    if (t == 0) out[b] = 1.f / (1.f + __expf(-(red[0] + bo[0])));
}

// ---------------------------------------------------------------------------
extern "C" void kernel_launch(void* const* d_in, const int* in_sizes, int n_in,
                              void* d_out, int out_size, void* d_ws, size_t ws_size,
                              hipStream_t stream)
{
    (void)in_sizes; (void)n_in; (void)out_size; (void)ws_size;
    const float* dense_x   = (const float*)d_in[0];
    const int*   sparse_ix = (const int*)  d_in[1];
    const float* gamma     = (const float*)d_in[2];
    const float* beta      = (const float*)d_in[3];
    const float* bn_mean   = (const float*)d_in[4];
    const float* bn_var    = (const float*)d_in[5];
    const float* tables    = (const float*)d_in[6];
    const float* w_lin     = (const float*)d_in[7];
    const float* b_lin     = (const float*)d_in[8];
    const float* wq0 = (const float*)d_in[9];
    const float* wk0 = (const float*)d_in[10];
    const float* wv0 = (const float*)d_in[11];
    const float* cw0 = (const float*)d_in[12];
    const float* wq1 = (const float*)d_in[13];
    const float* wk1 = (const float*)d_in[14];
    const float* wv1 = (const float*)d_in[15];
    const float* cw1 = (const float*)d_in[16];
    const float* W1 = (const float*)d_in[17];
    const float* b1 = (const float*)d_in[18];
    const float* W2 = (const float*)d_in[19];
    const float* b2 = (const float*)d_in[20];
    const float* W3 = (const float*)d_in[21];
    const float* b3 = (const float*)d_in[22];
    const float* W4 = (const float*)d_in[23];
    const float* b4 = (const float*)d_in[24];
    const float* wo = (const float*)d_in[25];
    const float* bo = (const float*)d_in[26];

    char* ws = (char*)d_ws;
    size_t off = 0;
    auto alloc = [&](size_t bytes) -> void* {
        void* p = ws + off;
        off = (off + bytes + 255) & ~(size_t)255;
        return p;
    };
    float*          emb    = (float*)         alloc((size_t)B_SZ * 416 * 4);
    unsigned short* h0     = (unsigned short*)alloc((size_t)B_SZ * DPAD * 2);
    float*          linout = (float*)         alloc((size_t)B_SZ * 4);
    float*          r1     = (float*)         alloc((size_t)B_SZ * 2048 * 4);
    float*          r2     = (float*)         alloc((size_t)B_SZ * 2048 * 4);
    float*          cino   = (float*)         alloc((size_t)B_SZ * 256 * 4);
    unsigned short* h1     = (unsigned short*)alloc((size_t)B_SZ * HPAD * 2);
    unsigned short* h2     = (unsigned short*)alloc((size_t)B_SZ * HPAD * 2);
    float*          h3     = (float*)         alloc((size_t)B_SZ * HID * 4);
    unsigned short* cw0p   = (unsigned short*)alloc((size_t)26 * 1  * 8 * 512 * 2);
    unsigned short* cw1p   = (unsigned short*)alloc((size_t)26 * 4  * 8 * 512 * 2);
    unsigned short* W1p    = (unsigned short*)alloc((size_t)14 * 25 * 512 * 2);
    unsigned short* W2p    = (unsigned short*)alloc((size_t)13 * 25 * 512 * 2);
    unsigned short* W3p    = (unsigned short*)alloc((size_t)13 * 25 * 512 * 2);

    // --- weight packing into WMMA B-fragment layout ---
    pack_b_tiles<<<26 * 1 * 8, 32, 0, stream>>>(cw0, cw0p, 26, 26, 128, 1, 8);
    pack_b_tiles<<<26 * 4 * 8, 32, 0, stream>>>(cw1, cw1p, 26, 128, 128, 4, 8);
    pack_b_tiles<<<14 * 25,    32, 0, stream>>>(W1, W1p, 1, DIN, HID, 14, 25);
    pack_b_tiles<<<13 * 25,    32, 0, stream>>>(W2, W2p, 1, HID, HID, 13, 25);
    pack_b_tiles<<<13 * 25,    32, 0, stream>>>(W3, W3p, 1, HID, HID, 13, 25);

    // --- embedding / BN / linear head ---
    embed_bn_kernel<<<B_SZ, 64, 0, stream>>>(dense_x, sparse_ix, gamma, beta,
                                             bn_mean, bn_var, tables, w_lin, b_lin,
                                             emb, h0, linout);

    // --- CIN layers (WMMA) ---
    cin_layer1<<<B_SZ, 256, 0, stream>>>(emb, wq0, wk0, wv0, cw0p, r1);
    cin_layer2<<<B_SZ, 256, 0, stream>>>(emb, r1, wq1, wk1, wv1, cw1p, r2);
    cin_sum<<<B_SZ, 256, 0, stream>>>(r1, r2, cino);

    // --- MLP (WMMA GEMMs) ---
    zero_pad_cols<<<(B_SZ * (HPAD - HID) + 255) / 256, 256, 0, stream>>>(h1, HPAD, HID);
    zero_pad_cols<<<(B_SZ * (HPAD - HID) + 255) / 256, 256, 0, stream>>>(h2, HPAD, HID);
    dim3 g1(B_SZ / 16, 25);
    gemm_bf16_wmma<<<g1, 32, 0, stream>>>(h0, DPAD, W1p, b1, nullptr, 0, h1, HPAD, 14, 25, 1);
    gemm_bf16_wmma<<<g1, 32, 0, stream>>>(h1, HPAD, W2p, b2, nullptr, 0, h2, HPAD, 13, 25, 1);
    gemm_bf16_wmma<<<g1, 32, 0, stream>>>(h2, HPAD, W3p, b3, h3, HID, nullptr, 0, 13, 25, 1);

    // --- final head ---
    final_kernel<<<B_SZ, 256, 0, stream>>>(h3, W4, b4, linout, cino, wo, bo,
                                           (float*)d_out);
}
